// Spatial_Aggregation_18717467476252
// MI455X (gfx1250) — matrix operand order repaired and checked by reading, not compiled, not run on hardware
//
#include <hip/hip_runtime.h>

// ---------------------------------------------------------------------------
// Spatial_Aggregation fused attention kernel for gfx1250 (MI455X, wave32)
// All GEMMs via v_wmma_f32_16x16x32_bf16; softmax in fp32; slice staged in LDS.
// ---------------------------------------------------------------------------

typedef __attribute__((ext_vector_type(16))) __bf16 v16bf;
typedef __attribute__((ext_vector_type(8)))  __bf16 v8bf;
typedef __attribute__((ext_vector_type(8)))  float  v8f;

#define BB 16
#define NN 512
#define TT 24
#define FF 64
#define SLICES   (BB * TT)        // 384
#define RBLK     16               // P-rows per workgroup
#define NBLKS    (NN / RBLK)      // 32
#define HC       (2 * FF)         // 128 (concat width)
#define STRIDE_B (NN * TT * FF)   // 786432
#define STRIDE_N (TT * FF)        // 1536
#define OUT_HALF ((size_t)BB * NN * TT * FF) // 12582912 floats per output tensor

// LDS layout (bytes); every region is 32B aligned, total = 300 KB of 320 KB WGP LDS
#define OFF_H   0                          // h   (512 x 128 bf16) row-major
#define OFF_XT  (OFF_H  + NN * HC * 2)     // x^T ( 64 x 512 bf16)
#define OFF_UT  (OFF_XT + FF * NN * 2)     // u^T ( 64 x 512 bf16)
#define OFF_W   (OFF_UT + FF * NN * 2)     // W1  ( 64 x  64 bf16) row-major
#define OFF_S   (OFF_W  + FF * FF * 2)     // S/P ( 16 x 512 f32)
#define OFF_O   (OFF_S  + RBLK * NN * 4)   // O   ( 16 x 128 bf16) [x-half | u-half]
#define SMEM_BYTES (OFF_O + RBLK * HC * 2) // 307200

__device__ __forceinline__ v16bf pack16(v8bf lo, v8bf hi) {
    v16bf a;
#pragma unroll
    for (int e = 0; e < 8; ++e) { a[e] = lo[e]; a[8 + e] = hi[e]; }
    return a;
}

__global__ __launch_bounds__(256, 1)
void spag_attn_kernel(const float* __restrict__ ori, const float* __restrict__ unc,
                      const float* __restrict__ emb, const float* __restrict__ W1,
                      float* __restrict__ out)
{
    extern __shared__ __align__(128) char smem[];
    __bf16* sh_h  = (__bf16*)(smem + OFF_H);
    __bf16* sh_xT = (__bf16*)(smem + OFF_XT);
    __bf16* sh_uT = (__bf16*)(smem + OFF_UT);
    __bf16* sh_W  = (__bf16*)(smem + OFF_W);
    float*  sh_S  = (float*) (smem + OFF_S);
    __bf16* sh_O  = (__bf16*)(smem + OFF_O);

    const int tid = threadIdx.x;
    const int blk = blockIdx.x;          // row block 0..31
    const int s   = blockIdx.y;          // slice = b*T + t
    const int b   = s / TT;
    const int t   = s - b * TT;

    const float* xg = ori + (size_t)b * STRIDE_B + (size_t)t * FF;
    const float* ug = unc + (size_t)b * STRIDE_B + (size_t)t * FF;

    // ---------------- load + bf16 convert + transpose-stage ----------------
#pragma unroll 4
    for (int it = 0; it < 32; ++it) {              // 512*64 floats / (256thr*4)
        int idx = tid + it * 256;                  // float4 index
        int n = idx >> 4;
        int f = (idx & 15) << 2;
        float4 vx = *(const float4*)(xg + (size_t)n * STRIDE_N + f);
        float4 vu = *(const float4*)(ug + (size_t)n * STRIDE_N + f);
        float4 ve = *(const float4*)(emb + n * FF + f);
        __bf16* hrow = sh_h + n * HC;
        hrow[f + 0] = (__bf16)vx.x; hrow[f + 1] = (__bf16)vx.y;
        hrow[f + 2] = (__bf16)vx.z; hrow[f + 3] = (__bf16)vx.w;
        hrow[FF + f + 0] = (__bf16)ve.x; hrow[FF + f + 1] = (__bf16)ve.y;
        hrow[FF + f + 2] = (__bf16)ve.z; hrow[FF + f + 3] = (__bf16)ve.w;
        sh_xT[(f + 0) * NN + n] = (__bf16)vx.x; sh_xT[(f + 1) * NN + n] = (__bf16)vx.y;
        sh_xT[(f + 2) * NN + n] = (__bf16)vx.z; sh_xT[(f + 3) * NN + n] = (__bf16)vx.w;
        sh_uT[(f + 0) * NN + n] = (__bf16)vu.x; sh_uT[(f + 1) * NN + n] = (__bf16)vu.y;
        sh_uT[(f + 2) * NN + n] = (__bf16)vu.z; sh_uT[(f + 3) * NN + n] = (__bf16)vu.w;
    }
#pragma unroll
    for (int it = 0; it < 4; ++it) {               // W1: 64x64
        int idx = tid + it * 256;
        int j = idx >> 4;
        int f = (idx & 15) << 2;
        float4 vw = *(const float4*)(W1 + j * FF + f);
        sh_W[j * FF + f + 0] = (__bf16)vw.x; sh_W[j * FF + f + 1] = (__bf16)vw.y;
        sh_W[j * FF + f + 2] = (__bf16)vw.z; sh_W[j * FF + f + 3] = (__bf16)vw.w;
    }
    __syncthreads();

    const int w  = tid >> 5;   // wave 0..7
    const int l  = tid & 31;
    const int r  = l & 15;
    const int hi = l >> 4;

    // ---------------- stage 1: S = relu(h_blk @ h_all^T) -------------------
    {
        // A fragments for this 16-row block (K = 128 -> 4 chunks of 32)
        v16bf afrag[4];
#pragma unroll
        for (int kk = 0; kk < 4; ++kk) {
            const __bf16* ap = sh_h + (blk * RBLK + r) * HC + kk * 32 + hi * 16;
            afrag[kk] = pack16(*(const v8bf*)ap, *(const v8bf*)(ap + 8));
        }
#pragma unroll
        for (int jj = 0; jj < 4; ++jj) {           // 32 col tiles / 8 waves
            int j = w * 4 + jj;
            v8f acc = {};
#pragma unroll
            for (int kk = 0; kk < 4; ++kk) {
                const __bf16* bp = sh_h + (j * 16 + r) * HC + kk * 32 + hi * 16;
                v16bf bf = *(const v16bf*)bp;
                acc = __builtin_amdgcn_wmma_f32_16x16x32_bf16(
                          false, afrag[kk], false, bf, (short)0, acc, false, false);
            }
#pragma unroll
            for (int v = 0; v < 8; ++v) {
                float val = acc[v];
                sh_S[(v + 8 * hi) * NN + j * 16 + r] = val > 0.f ? val : 0.f;
            }
        }
    }
    __syncthreads();

    // ---------------- softmax over 512 cols (fp32) -------------------------
    {
        int row = tid >> 4;                        // 16 threads per row
        int sub = tid & 15;
        float* p = sh_S + row * NN;
        float vals[32];
        float m = -3.0e38f;
#pragma unroll
        for (int k = 0; k < 32; ++k) { vals[k] = p[sub + 16 * k]; m = fmaxf(m, vals[k]); }
#pragma unroll
        for (int off = 8; off > 0; off >>= 1) m = fmaxf(m, __shfl_xor(m, off, 16));
        float ssum = 0.f;
#pragma unroll
        for (int k = 0; k < 32; ++k) { vals[k] = __expf(vals[k] - m); ssum += vals[k]; }
#pragma unroll
        for (int off = 8; off > 0; off >>= 1) ssum += __shfl_xor(ssum, off, 16);
        float inv = 1.0f / ssum;
#pragma unroll
        for (int k = 0; k < 32; ++k) p[sub + 16 * k] = vals[k] * inv;
    }
    __syncthreads();

    // ---------------- stage 2: O = P @ {x,u}  (K = 512) --------------------
    {
        int mat = w & 1;                           // 0 -> x, 1 -> u
        int jc  = w >> 1;                          // feature tile 0..3
        const __bf16* bt = mat ? sh_uT : sh_xT;
        v8f acc = {};
        for (int kk = 0; kk < 16; ++kk) {
            const float* ap = sh_S + r * NN + kk * 32 + hi * 16;
            float4 a0 = *(const float4*)(ap);
            float4 a1 = *(const float4*)(ap + 4);
            float4 a2 = *(const float4*)(ap + 8);
            float4 a3 = *(const float4*)(ap + 12);
            v16bf a;
            a[0]  = (__bf16)a0.x; a[1]  = (__bf16)a0.y; a[2]  = (__bf16)a0.z; a[3]  = (__bf16)a0.w;
            a[4]  = (__bf16)a1.x; a[5]  = (__bf16)a1.y; a[6]  = (__bf16)a1.z; a[7]  = (__bf16)a1.w;
            a[8]  = (__bf16)a2.x; a[9]  = (__bf16)a2.y; a[10] = (__bf16)a2.z; a[11] = (__bf16)a2.w;
            a[12] = (__bf16)a3.x; a[13] = (__bf16)a3.y; a[14] = (__bf16)a3.z; a[15] = (__bf16)a3.w;
            const __bf16* bp = bt + (jc * 16 + r) * NN + kk * 32 + hi * 16;
            v16bf bf = *(const v16bf*)bp;
            acc = __builtin_amdgcn_wmma_f32_16x16x32_bf16(
                      false, a, false, bf, (short)0, acc, false, false);
        }
#pragma unroll
        for (int v = 0; v < 8; ++v)
            sh_O[(v + 8 * hi) * HC + mat * FF + jc * 16 + r] = (__bf16)acc[v];
    }
    __syncthreads();

    // ---------------- stage 3: Y = relu(O @ W1^T), store -------------------
    {
        int mat = w & 1;
        int jb  = w >> 1;                          // output feature tile 0..3
        v8f acc = {};
#pragma unroll
        for (int kk = 0; kk < 2; ++kk) {
            const __bf16* ap = sh_O + r * HC + mat * FF + kk * 32 + hi * 16;
            v16bf a = pack16(*(const v8bf*)ap, *(const v8bf*)(ap + 8));
            const __bf16* bp = sh_W + (jb * 16 + r) * FF + kk * 32 + hi * 16;
            v16bf bf = *(const v16bf*)bp;
            acc = __builtin_amdgcn_wmma_f32_16x16x32_bf16(
                      false, a, false, bf, (short)0, acc, false, false);
        }
        float* op = out + (mat ? OUT_HALF : 0) + (size_t)b * STRIDE_B + (size_t)t * FF;
#pragma unroll
        for (int v = 0; v < 8; ++v) {
            int n = blk * RBLK + v + 8 * hi;
            int f = jb * 16 + r;
            float y = acc[v];
            op[(size_t)n * STRIDE_N + f] = y > 0.f ? y : 0.f;
        }
    }
}

extern "C" void kernel_launch(void* const* d_in, const int* in_sizes, int n_in,
                              void* d_out, int out_size, void* d_ws, size_t ws_size,
                              hipStream_t stream) {
    (void)in_sizes; (void)n_in; (void)d_ws; (void)ws_size; (void)out_size;
    const float* ori = (const float*)d_in[0];
    const float* unc = (const float*)d_in[1];
    const float* emb = (const float*)d_in[2];
    const float* W1  = (const float*)d_in[3];
    float* out = (float*)d_out;

    dim3 grid(NBLKS, SLICES);   // 32 row-blocks x 384 slices
    dim3 block(256);            // 8 wave32s
    hipLaunchKernelGGL(spag_attn_kernel, grid, block, SMEM_BYTES, stream,
                       ori, unc, emb, W1, out);
}
